// Attention_14645838479453
// MI455X (gfx1250) — matrix operand order repaired and checked by reading, not compiled, not run on hardware
//
#include <hip/hip_runtime.h>

typedef __attribute__((ext_vector_type(16))) __bf16 v16bf;
typedef __attribute__((ext_vector_type(8)))  float  v8f;

#define Bb 2
#define Nn 2048
#define Cc 768
#define Hh 12
#define Dd 64
#define ATT_SCALE 0.125f     // 1/sqrt(64)
#define PSTRIDE  2052        // fp32 LDS row stride (bank-rotating pad)
#define PBSTRIDE 4104        // bf16 view of the same stripe (2*PSTRIDE)

union BFrag {
  v16bf  v;
  __bf16 h[16];
  uint4  q[2];
};
union BHalf {
  __bf16 h[8];
  uint4  q;
};

static __device__ inline __bf16 f2bf(float f) { return static_cast<__bf16>(f); }

// elements 0..7 from p0, 8..15 from p1 (bf16 direct, 2x b128)
static __device__ inline void load16bf(BFrag& fr, const __bf16* p0, const __bf16* p1) {
  fr.q[0] = *(const uint4*)(p0);
  fr.q[1] = *(const uint4*)(p1);
}

static __device__ inline v8f wmma_bf16(const BFrag& a, const BFrag& b, v8f c) {
  return __builtin_amdgcn_wmma_f32_16x16x32_bf16(false, a.v, false, b.v, (short)0, c, false, false);
}

// ---------------------------------------------------------------------------
// Kernel 0: fp32 -> bf16 streaming convert (n8 = element count / 8)
// ---------------------------------------------------------------------------
__global__ __launch_bounds__(256) void cvt_kernel(
    const float* __restrict__ src, __bf16* __restrict__ dst, int n8)
{
  const int i = blockIdx.x * 256 + threadIdx.x;
  if (i >= n8) return;
  float4 a = ((const float4*)src)[2 * i];
  float4 b = ((const float4*)src)[2 * i + 1];
  BHalf st;
  st.h[0] = f2bf(a.x); st.h[1] = f2bf(a.y); st.h[2] = f2bf(a.z); st.h[3] = f2bf(a.w);
  st.h[4] = f2bf(b.x); st.h[5] = f2bf(b.y); st.h[6] = f2bf(b.z); st.h[7] = f2bf(b.w);
  ((uint4*)dst)[i] = st.q;
}

// ---------------------------------------------------------------------------
// Kernel 1: qkv = x @ qkv_w^T (all-bf16 operands) -> Q,K (B,H,N,HD); V^T (B,H,HD,N)
// grid (36, 64), block 256. Wave computes 16 rows x 32 cols (2 accumulators).
// ---------------------------------------------------------------------------
__global__ __launch_bounds__(256) void qkv_kernel(
    const __bf16* __restrict__ x, const __bf16* __restrict__ w,
    __bf16* __restrict__ Qb, __bf16* __restrict__ Kb, __bf16* __restrict__ Vt)
{
  const int lane = threadIdx.x & 31;
  const int wave = threadIdx.x >> 5;
  const int lrow = lane & 15;
  const int hi   = lane >> 4;
  const int m0 = blockIdx.y * 64 + (wave >> 1) * 16;
  const int n0 = blockIdx.x * 64 + (wave & 1) * 32;

  v8f acc0 = {}; v8f acc1 = {};
  const __bf16* arow = x + (size_t)(m0 + lrow) * Cc;
  const __bf16* brow0 = w + (size_t)(n0 + lrow) * Cc;
  const __bf16* brow1 = w + (size_t)(n0 + 16 + lrow) * Cc;
  for (int kk = 0; kk < Cc; kk += 32) {
    BFrag af;  const __bf16* ap = arow + kk + hi * 16;
    load16bf(af, ap, ap + 8);
    BFrag b0;  const __bf16* bp0 = brow0 + kk + hi * 16;
    load16bf(b0, bp0, bp0 + 8);
    acc0 = wmma_bf16(af, b0, acc0);
    BFrag b1;  const __bf16* bp1 = brow1 + kk + hi * 16;
    load16bf(b1, bp1, bp1 + 8);
    acc1 = wmma_bf16(af, b1, acc1);
  }

  const int bidx  = m0 >> 11;
  const int nbase = (m0 & (Nn - 1)) + hi * 8;
  for (int j = 0; j < 2; ++j) {
    v8f acc = j ? acc1 : acc0;
    const int r  = n0 + j * 16 + lrow;     // uniform t,h per 16-col tile
    const int t  = r / Cc;
    const int rr = r - t * Cc;
    const int h  = rr >> 6;
    const int d  = rr & 63;
    if (t == 2) {                          // V^T: contiguous in n -> single b128 store
      BHalf st;
      for (int i = 0; i < 8; ++i) st.h[i] = f2bf(acc[i]);
      *(uint4*)(Vt + ((size_t)(bidx * Hh + h) * Dd + d) * Nn + nbase) = st.q;
    } else {                               // Q/K: (B,H,N,HD)
      __bf16* dst = (t == 0 ? Qb : Kb) + ((size_t)(bidx * Hh + h) * Nn + nbase) * Dd + d;
      for (int i = 0; i < 8; ++i) dst[(size_t)i * Dd] = f2bf(acc[i]);
    }
  }
}

// ---------------------------------------------------------------------------
// Kernel 2: per (b, 32-query block): loop h: S = Q K^T * scale (WMMA -> LDS f32),
// row softmax (+score accumulate, in-place bf16 pack of P), O = P V (WMMA, P from LDS).
// grid (128), block 256, dynamic LDS 32*PSTRIDE*4 bytes.
// ---------------------------------------------------------------------------
__global__ __launch_bounds__(256) void attn_kernel(
    const __bf16* __restrict__ Qb, const __bf16* __restrict__ Kb,
    const __bf16* __restrict__ Vt, __bf16* __restrict__ AO,
    float* __restrict__ score)
{
  extern __shared__ char smem[];
  float*  P   = (float*)smem;        // [32][PSTRIDE] fp32 view
  __bf16* Pbf = (__bf16*)smem;       // [32][PBSTRIDE] bf16 view (same stripes)

  const int lane = threadIdx.x & 31;
  const int wave = threadIdx.x >> 5;
  const int lrow = lane & 15;
  const int hi   = lane >> 4;
  const int b  = blockIdx.x >> 6;
  const int q0 = (blockIdx.x & 63) * 32;
  const int mt = wave >> 2;          // PV: row tile 0..1
  const int nt = wave & 3;           // PV: col tile 0..3

  for (int h = 0; h < Hh; ++h) {
    const __bf16* Qh = Qb + (size_t)(b * Hh + h) * Nn * Dd;
    const __bf16* Kh = Kb + (size_t)(b * Hh + h) * Nn * Dd;
    const __bf16* Vh = Vt + (size_t)(b * Hh + h) * Dd * Nn;

    // Q fragments: 2 M-tiles x 2 K-steps, kept in registers
    BFrag qf[2][2];
    for (int m = 0; m < 2; ++m)
      for (int ks = 0; ks < 2; ++ks) {
        const __bf16* ap = Qh + (size_t)(q0 + m * 16 + lrow) * Dd + ks * 32 + hi * 16;
        load16bf(qf[m][ks], ap, ap + 8);
      }

    // S block: each wave sweeps 16 of the 128 key-column tiles
    for (int jt = wave; jt < Nn / 16; jt += 8) {
      const int n0 = jt * 16;
      if (jt + 8 < Nn / 16)
        __builtin_prefetch(Kh + (size_t)((jt + 8) * 16 + lrow) * Dd, 0, 1);
      v8f s0 = {}; v8f s1 = {};
      for (int ks = 0; ks < 2; ++ks) {
        const __bf16* bp = Kh + (size_t)(n0 + lrow) * Dd + ks * 32 + hi * 16;
        BFrag kf; load16bf(kf, bp, bp + 8);
        s0 = wmma_bf16(qf[0][ks], kf, s0);
        s1 = wmma_bf16(qf[1][ks], kf, s1);
      }
      float* pc = P + n0 + lrow;
      for (int i = 0; i < 8; ++i) {
        pc[(size_t)(hi * 8 + i) * PSTRIDE]      = s0[i] * ATT_SCALE;
        pc[(size_t)(16 + hi * 8 + i) * PSTRIDE] = s1[i] * ATT_SCALE;
      }
    }
    __syncthreads();

    // softmax per row (wave owns 4 rows); score accumulate; pack P to bf16 in place
    for (int r = 0; r < 4; ++r) {
      const int row = wave * 4 + r;
      float*  pr  = P   + (size_t)row * PSTRIDE;
      __bf16* pbr = Pbf + (size_t)row * PBSTRIDE;
      float mx = -3.4e38f;
      for (int c = lane; c < Nn; c += 32) mx = fmaxf(mx, pr[c]);
      for (int off = 16; off > 0; off >>= 1) mx = fmaxf(mx, __shfl_xor(mx, off, 32));
      float s = 0.f;
      for (int c = lane; c < Nn; c += 32) { float e = __expf(pr[c] - mx); pr[c] = e; s += e; }
      for (int off = 16; off > 0; off >>= 1) s += __shfl_xor(s, off, 32);
      const float inv  = 1.f / s;
      const float invH = inv * (1.f / (float)Hh);
      float* sc = score + (size_t)b * Nn * Nn + (size_t)(q0 + row) * Nn;
      // front-to-back: bf16 write at byte 2c never overtakes fp32 reads at byte 4c
      if (h == 0) {
        for (int c = lane; c < Nn; c += 32) {
          float e = pr[c];
          pbr[c] = f2bf(e * inv);
          sc[c]  = e * invH;
          asm volatile("" ::: "memory");
        }
      } else {
        for (int c = lane; c < Nn; c += 32) {
          float e = pr[c];
          pbr[c] = f2bf(e * inv);
          sc[c] += e * invH;
          asm volatile("" ::: "memory");
        }
      }
    }
    __syncthreads();

    // O = P (32x2048 bf16 in LDS) x V (2048x64); wave owns one 16x16 tile
    v8f o = {};
    for (int kk = 0; kk < Nn; kk += 32) {
      const __bf16* pp = Pbf + (size_t)(mt * 16 + lrow) * PBSTRIDE + kk + hi * 16;
      BFrag af; load16bf(af, pp, pp + 8);
      const __bf16* vp = Vh + (size_t)(nt * 16 + lrow) * Nn + kk + hi * 16;
      BFrag vf; load16bf(vf, vp, vp + 8);
      o = wmma_bf16(af, vf, o);
    }
    __bf16* op = AO + ((size_t)b * Nn + (q0 + mt * 16 + hi * 8)) * Cc + h * Dd + nt * 16 + lrow;
    for (int i = 0; i < 8; ++i) op[(size_t)i * Cc] = f2bf(o[i]);
    __syncthreads();   // protect P before next head overwrites it
  }
}

// ---------------------------------------------------------------------------
// Kernel 3: out = attn_out @ proj_w^T + proj_b  (bf16 operands, fp32 result)
// grid (12, 64), block 256.
// ---------------------------------------------------------------------------
__global__ __launch_bounds__(256) void proj_kernel(
    const __bf16* __restrict__ a, const __bf16* __restrict__ w,
    const float* __restrict__ bias, float* __restrict__ out)
{
  const int lane = threadIdx.x & 31;
  const int wave = threadIdx.x >> 5;
  const int lrow = lane & 15;
  const int hi   = lane >> 4;
  const int m0 = blockIdx.y * 64 + (wave >> 1) * 16;
  const int n0 = blockIdx.x * 64 + (wave & 1) * 32;

  v8f acc0 = {}; v8f acc1 = {};
  const __bf16* arow  = a + (size_t)(m0 + lrow) * Cc;
  const __bf16* brow0 = w + (size_t)(n0 + lrow) * Cc;
  const __bf16* brow1 = w + (size_t)(n0 + 16 + lrow) * Cc;
  for (int kk = 0; kk < Cc; kk += 32) {
    BFrag af;  const __bf16* ap = arow + kk + hi * 16;
    load16bf(af, ap, ap + 8);
    BFrag b0;  const __bf16* bp0 = brow0 + kk + hi * 16;
    load16bf(b0, bp0, bp0 + 8);
    acc0 = wmma_bf16(af, b0, acc0);
    BFrag b1;  const __bf16* bp1 = brow1 + kk + hi * 16;
    load16bf(b1, bp1, bp1 + 8);
    acc1 = wmma_bf16(af, b1, acc1);
  }
  for (int j = 0; j < 2; ++j) {
    v8f acc = j ? acc1 : acc0;
    const int r = n0 + j * 16 + lrow;
    const float bv = bias[r];
    float* op = out + (size_t)(m0 + hi * 8) * Cc + r;
    for (int i = 0; i < 8; ++i) op[(size_t)i * Cc] = acc[i] + bv;
  }
}

extern "C" void kernel_launch(void* const* d_in, const int* in_sizes, int n_in,
                              void* d_out, int out_size, void* d_ws, size_t ws_size,
                              hipStream_t stream) {
  (void)in_sizes; (void)n_in; (void)out_size; (void)ws_size;
  const float* x      = (const float*)d_in[0];
  const float* qkv_w  = (const float*)d_in[1];
  const float* proj_w = (const float*)d_in[2];
  const float* proj_b = (const float*)d_in[3];

  float* out   = (float*)d_out;                     // (B,N,C) fp32
  float* score = out + (size_t)Bb * Nn * Cc;        // (B,N,N) fp32

  const size_t e  = (size_t)Bb * Hh * Nn * Dd;      // per-tensor bf16 elements
  const size_t nx = (size_t)Bb * Nn * Cc;           // x elements
  const size_t nq = (size_t)3 * Cc * Cc;            // qkv_w elements
  const size_t np = (size_t)Cc * Cc;                // proj_w elements

  __bf16* Qb  = (__bf16*)d_ws;
  __bf16* Kb  = Qb  + e;
  __bf16* Vt  = Kb  + e;
  __bf16* AO  = Vt  + e;            // (B,N,C) bf16 attention output
  __bf16* xb  = AO  + e;            // bf16 copies of fp32 inputs
  __bf16* qwb = xb  + nx;
  __bf16* pwb = qwb + nq;

  const int g_x = (int)(nx / 8), g_q = (int)(nq / 8), g_p = (int)(np / 8);
  cvt_kernel<<<dim3((g_x + 255) / 256), dim3(256), 0, stream>>>(x,      xb,  g_x);
  cvt_kernel<<<dim3((g_q + 255) / 256), dim3(256), 0, stream>>>(qkv_w,  qwb, g_q);
  cvt_kernel<<<dim3((g_p + 255) / 256), dim3(256), 0, stream>>>(proj_w, pwb, g_p);

  qkv_kernel<<<dim3(36, 64), dim3(256), 0, stream>>>(xb, qwb, Qb, Kb, Vt);
  attn_kernel<<<dim3(Bb * (Nn / 32)), dim3(256), 32 * PSTRIDE * sizeof(float), stream>>>(
      Qb, Kb, Vt, AO, score);
  proj_kernel<<<dim3(Cc / 64, (Bb * Nn) / 64), dim3(256), 0, stream>>>(AO, pwb, proj_b, out);
}
